// TrajectoryGenerator_10411000725934
// MI455X (gfx1250) — compile-verified, matrix operand order checked by reading, not run on hardware
//
#include <hip/hip_runtime.h>

typedef __bf16 bf16;
typedef __attribute__((ext_vector_type(16))) __bf16 v16bf;
typedef __attribute__((ext_vector_type(8)))  float  v8f;
typedef __attribute__((ext_vector_type(4)))  unsigned int v4u;
typedef __attribute__((ext_vector_type(8)))  int v8i;
typedef __attribute__((ext_vector_type(4)))  int v4i;

#define BN_INV 0.99999500003749969f
#define NPED 1024
#define HDIM 128
#define EDIM 64
#define KLSTM 192   // E + H
#define GATES 512   // 4*H
#define B1 512
#define B2 1024
#define M1 1024
#define KM1 1152    // H + B2
#define TSTEPS 12

// ---------------------------------------------------------------------------
// WMMA helpers
// ---------------------------------------------------------------------------
__device__ __forceinline__ v8f wmma_bf16(v16bf a, v16bf b, v8f c) {
  return __builtin_amdgcn_wmma_f32_16x16x32_bf16(false, a, false, b, (short)0, c,
                                                 false, false);
}

// Load a 16x32 bf16 fragment (A-layout; also valid for B when the weight is
// stored [Nout x K] row-major, since lane n then holds W row n contiguously).
// ISA layout: lanes 0-15 -> rows, K {0..7, 16..23}; lanes 16-31 -> K {8..15, 24..31}.
template <typename PT>
__device__ __forceinline__ v16bf frag_rm(PT X, int ld, int row0, int k0) {
  const int lane = threadIdx.x & 31;
  PT p = X + (size_t)(row0 + (lane & 15)) * ld + k0 + ((lane >> 4) << 3);
  v16bf f;
#pragma unroll
  for (int e = 0; e < 8; ++e) f[e] = p[e];
#pragma unroll
  for (int e = 0; e < 8; ++e) f[e + 8] = p[e + 16];
  return f;
}

__device__ __forceinline__ float sigmoidf_(float x) { return 1.0f / (1.0f + expf(-x)); }

// ---------------------------------------------------------------------------
// Weight prep / init kernels
// ---------------------------------------------------------------------------
__global__ void f2bf_kernel(const float* __restrict__ src, bf16* __restrict__ dst, int n) {
  for (int i = blockIdx.x * blockDim.x + threadIdx.x; i < n; i += gridDim.x * blockDim.x)
    dst[i] = (bf16)src[i];
}

__global__ void build_lstm_w(const float* __restrict__ Wih, const float* __restrict__ Whh,
                             const float* __restrict__ bih, const float* __restrict__ bhh,
                             bf16* __restrict__ Wg, float* __restrict__ bias) {
  const int stride = gridDim.x * blockDim.x;
  for (int i = blockIdx.x * blockDim.x + threadIdx.x; i < GATES * KLSTM; i += stride) {
    int r = i / KLSTM, c = i % KLSTM;
    float v = (c < EDIM) ? Wih[r * EDIM + c] : Whh[r * HDIM + (c - EDIM)];
    Wg[i] = (bf16)v;
  }
  for (int i = blockIdx.x * blockDim.x + threadIdx.x; i < GATES; i += stride)
    bias[i] = bih[i] + bhh[i];
}

__global__ void init_state(const float* __restrict__ last_pos,
                           const float* __restrict__ last_pos_rel,
                           const float* __restrict__ h0, const float* __restrict__ c0,
                           const float* __restrict__ emb_W, const float* __restrict__ emb_b,
                           bf16* __restrict__ xg, float* __restrict__ c,
                           float* __restrict__ curr) {
  const int stride = gridDim.x * blockDim.x;
  for (int i = blockIdx.x * blockDim.x + threadIdx.x; i < NPED * KLSTM; i += stride) {
    int r = i / KLSTM, cc = i % KLSTM;
    float v;
    if (cc < EDIM)
      v = last_pos_rel[r * 2 + 0] * emb_W[cc * 2 + 0] +
          last_pos_rel[r * 2 + 1] * emb_W[cc * 2 + 1] + emb_b[cc];
    else
      v = h0[r * HDIM + (cc - EDIM)];
    xg[i] = (bf16)v;
  }
  for (int i = blockIdx.x * blockDim.x + threadIdx.x; i < NPED * HDIM; i += stride)
    c[i] = c0[i];
  for (int i = blockIdx.x * blockDim.x + threadIdx.x; i < NPED * 2; i += stride)
    curr[i] = last_pos[i];
}

// ---------------------------------------------------------------------------
// Generic bf16 WMMA GEMM:  out[M x N] = epi(A[M x K] * W[N x K]^T)
// grid = (N/128, M/32), block = 256 (8 waves). Wave w -> 16 cols, 32 rows.
// EPI: 1 = +bias (f32 out), 2 = relu(bn(+bias)) ; OBF: bf16 output
// ---------------------------------------------------------------------------
template <int EPI, bool OBF>
__global__ __launch_bounds__(256) void gemm_bf16(
    const bf16* __restrict__ A, int lda, const bf16* __restrict__ W, int ldw,
    const float* __restrict__ bias, const float* __restrict__ g,
    const float* __restrict__ be, float* __restrict__ outF, bf16* __restrict__ outB,
    int ldo, int K) {
  const int wave = threadIdx.x >> 5, lane = threadIdx.x & 31;
  const int n0 = blockIdx.x * 128 + wave * 16;
  const int m0 = blockIdx.y * 32;
  v8f acc0 = {}, acc1 = {};
  for (int k0 = 0; k0 < K; k0 += 32) {
    if (k0 + 32 < K)  // gfx1250 global_prefetch_b8 on the weight stream
      __builtin_prefetch((const void*)(W + (size_t)(n0 + (lane & 15)) * ldw + k0 + 32), 0, 0);
    v16bf a0 = frag_rm(A, lda, m0, k0);
    v16bf a1 = frag_rm(A, lda, m0 + 16, k0);
    v16bf b = frag_rm(W, ldw, n0, k0);
    acc0 = wmma_bf16(a0, b, acc0);
    acc1 = wmma_bf16(a1, b, acc1);
  }
  const int col = n0 + (lane & 15);
  const int mb = (lane >> 4) << 3;
  float bb = bias[col];
  float gg = (EPI == 2) ? g[col] * BN_INV : 1.0f;
  float bv = (EPI == 2) ? be[col] : 0.0f;
#pragma unroll
  for (int v = 0; v < 8; ++v) {
    float t0 = acc0[v] + bb, t1 = acc1[v] + bb;
    if (EPI == 2) {
      t0 = fmaxf(t0 * gg + bv, 0.0f);
      t1 = fmaxf(t1 * gg + bv, 0.0f);
    }
    int r0 = m0 + mb + v, r1 = m0 + 16 + mb + v;
    if (OBF) {
      outB[(size_t)r0 * ldo + col] = (bf16)t0;
      outB[(size_t)r1 * ldo + col] = (bf16)t1;
    } else {
      outF[(size_t)r0 * ldo + col] = t0;
      outF[(size_t)r1 * ldo + col] = t1;
    }
  }
}

// ---------------------------------------------------------------------------
// LSTM pointwise: gates -> c2,h2 ; rel_pos = h2*pos_W^T+b ; curr/preds/din
// grid = 1024 (one block per ped), block = 128 (one thread per h unit)
// ---------------------------------------------------------------------------
__global__ __launch_bounds__(128) void lstm_pointwise(
    const float* __restrict__ gates, float* __restrict__ c, bf16* __restrict__ xm,
    bf16* __restrict__ xg, float* __restrict__ curr, float* __restrict__ preds_t,
    const float* __restrict__ pos_W, const float* __restrict__ pos_b,
    const float* __restrict__ emb_W, const float* __restrict__ emb_b) {
  const int row = blockIdx.x;
  const int hh = threadIdx.x;  // 0..127
  const float* gr = gates + (size_t)row * GATES;
  float ig = sigmoidf_(gr[hh]);
  float fg = sigmoidf_(gr[HDIM + hh]);
  float gg = tanhf(gr[2 * HDIM + hh]);
  float og = sigmoidf_(gr[3 * HDIM + hh]);
  float c2 = fg * c[row * HDIM + hh] + ig * gg;
  float h2 = og * tanhf(c2);
  c[row * HDIM + hh] = c2;
  xm[(size_t)row * KM1 + hh] = (bf16)h2;  // h2 -> first 128 cols of m1 input

  __shared__ float red0[HDIM], red1[HDIM], rp[2];
  red0[hh] = h2 * pos_W[hh];
  red1[hh] = h2 * pos_W[HDIM + hh];
  __syncthreads();
  for (int s = HDIM / 2; s > 0; s >>= 1) {
    if (hh < s) {
      red0[hh] += red0[hh + s];
      red1[hh] += red1[hh + s];
    }
    __syncthreads();
  }
  if (hh == 0) {
    float r0 = red0[0] + pos_b[0], r1 = red1[0] + pos_b[1];
    rp[0] = r0; rp[1] = r1;
    curr[row * 2 + 0] += r0;
    curr[row * 2 + 1] += r1;
    preds_t[row * 2 + 0] = r0;
    preds_t[row * 2 + 1] = r1;
  }
  __syncthreads();
  if (hh < EDIM) {
    float d = rp[0] * emb_W[hh * 2 + 0] + rp[1] * emb_W[hh * 2 + 1] + emb_b[hh];
    xg[(size_t)row * KLSTM + hh] = (bf16)d;  // next-step spatial embedding
  }
}

// ---------------------------------------------------------------------------
// Fused PoolHiddenNet: one block per (scene, ped i).
//  - h2 rows of the scene staged into LDS by the Tensor Data Mover (TDM),
//    padded so each 128-elem h2 row lands at xin[j*192+64].
//  - emb(normalized rel) computed in VALU into xin[j*192 .. +64)
//  - pp1 (192->512) and pp2 (512->1024) WMMA, BN+ReLU, max over j.
// block = 256 (8 waves)
// ---------------------------------------------------------------------------
__global__ __launch_bounds__(256) void pool_kernel(
    const float* __restrict__ curr, const bf16* __restrict__ xm_h2,
    const bf16* __restrict__ pp1w, const float* __restrict__ pp1_b,
    const float* __restrict__ pp1_g, const float* __restrict__ pp1_be,
    const bf16* __restrict__ pp2w, const float* __restrict__ pp2_b,
    const float* __restrict__ pp2_g, const float* __restrict__ pp2_be,
    const float* __restrict__ sp_W, const float* __restrict__ sp_b,
    bf16* __restrict__ xm_pool) {
  __shared__ bf16 xin[32 * KLSTM];  // 12 KB  [emb(64) | h2(128)] per row j
  __shared__ bf16 x1[32 * B1];      // 32 KB
  const int bi = blockIdx.x;        // s*32 + i
  const int s = bi >> 5;
  const int tid = threadIdx.x;
  const float cix = curr[bi * 2 + 0], ciy = curr[bi * 2 + 1];

  // ---- TDM: DMA the 32x128 bf16 h2 tile (row stride 1152) into padded LDS.
  // D# per CDNA5 ISA ch.8: 2D tile, data_size=2B, pad after every 64 DWORDs
  // (one 256B row) by 32 DWORDs (the 128B emb slot of the next xin row).
  if (tid == 0) {
    unsigned long long ga =
        (unsigned long long)(xm_h2 + (size_t)(s * 32) * KM1);
    unsigned int ldsa = (unsigned int)(unsigned long long)(&xin[EDIM]);
    v4u g0;
    g0[0] = 1u;                                   // count=1 (valid), user mode
    g0[1] = ldsa;                                 // lds_addr (bytes)
    g0[2] = (unsigned int)ga;                     // global_addr[31:0]
    g0[3] = (unsigned int)((ga >> 32) & 0x01FFFFFFu) | (2u << 30);  // [56:32]|type=2
    v8i g1;
    g1[0] = (int)((1u << 16) |                    // data_size = 2 bytes
                  (1u << 20) |                    // pad_enable
                  (5u << 22) |                    // pad_interval: 64 DWORDs
                  (31u << 25));                   // pad_amount: 32 DWORDs
    g1[1] = (int)(1152u << 16);                   // tensor_dim0[15:0] = 1152
    g1[2] = (int)(32u << 16);                     // tensor_dim1[15:0] = 32
    g1[3] = (int)(128u << 16);                    // tile_dim0 = 128
    g1[4] = (int)32;                              // tile_dim1 = 32, tile_dim2 = 0
    g1[5] = (int)1152;                            // tensor_dim0_stride = 1152
    g1[6] = 0;
    g1[7] = 0;
    v4i gz4 = {0, 0, 0, 0};
    v8i gz8 = {0, 0, 0, 0, 0, 0, 0, 0};
    __builtin_amdgcn_tensor_load_to_lds(g0, g1, gz4, gz4, gz8, 0);
    __builtin_amdgcn_s_wait_tensorcnt(0);
  }

  // ---- spatial embedding of normalized relative positions (VALU, overlaps TDM)
  for (int it = tid; it < 32 * EDIM; it += 256) {
    int j = it >> 6, e = it & 63;
    float rx = curr[(s * 32 + j) * 2 + 0] - cix;
    float ry = curr[(s * 32 + j) * 2 + 1] - ciy;
    float inv = 1.0f / fmaxf(sqrtf(rx * rx + ry * ry), 1e-12f);
    rx *= inv; ry *= inv;
    xin[j * KLSTM + e] = (bf16)(rx * sp_W[e * 2 + 0] + ry * sp_W[e * 2 + 1] + sp_b[e]);
  }
  __syncthreads();

  const int wave = tid >> 5, lane = tid & 31;
  const int mb = (lane >> 4) << 3;

  // ---- pp1 -> x1 (32 x 512) in LDS; each wave owns 64 cols (2x2 tile blocking)
  for (int nt = 0; nt < 4; nt += 2) {
    const int n0 = wave * 64 + nt * 16;
    const int n1 = n0 + 16;
    v8f a00 = {}, a01 = {}, a10 = {}, a11 = {};
#pragma unroll
    for (int k0 = 0; k0 < KLSTM; k0 += 32) {
      v16bf a0 = frag_rm(&xin[0], KLSTM, 0, k0);
      v16bf a1 = frag_rm(&xin[0], KLSTM, 16, k0);
      v16bf b0 = frag_rm(pp1w, KLSTM, n0, k0);
      v16bf b1 = frag_rm(pp1w, KLSTM, n1, k0);
      a00 = wmma_bf16(a0, b0, a00);
      a10 = wmma_bf16(a1, b0, a10);
      a01 = wmma_bf16(a0, b1, a01);
      a11 = wmma_bf16(a1, b1, a11);
    }
#pragma unroll
    for (int h = 0; h < 2; ++h) {
      const int col = (h ? n1 : n0) + (lane & 15);
      const v8f& q0 = h ? a01 : a00;
      const v8f& q1 = h ? a11 : a10;
      float bb = pp1_b[col], gg = pp1_g[col] * BN_INV, bv = pp1_be[col];
#pragma unroll
      for (int v = 0; v < 8; ++v) {
        x1[(mb + v) * B1 + col] = (bf16)fmaxf((q0[v] + bb) * gg + bv, 0.0f);
        x1[(16 + mb + v) * B1 + col] = (bf16)fmaxf((q1[v] + bb) * gg + bv, 0.0f);
      }
    }
  }
  __syncthreads();

  // ---- pp2 + BN + ReLU + max over the 32 neighbour rows (2x2 tile blocking)
  for (int nt = 0; nt < 8; nt += 2) {
    const int n0 = (wave * 8 + nt) * 16;
    const int n1 = n0 + 16;
    v8f a00 = {}, a01 = {}, a10 = {}, a11 = {};
#pragma unroll
    for (int k0 = 0; k0 < B1; k0 += 32) {
      v16bf a0 = frag_rm(&x1[0], B1, 0, k0);
      v16bf a1 = frag_rm(&x1[0], B1, 16, k0);
      v16bf b0 = frag_rm(pp2w, B1, n0, k0);
      v16bf b1 = frag_rm(pp2w, B1, n1, k0);
      a00 = wmma_bf16(a0, b0, a00);
      a10 = wmma_bf16(a1, b0, a10);
      a01 = wmma_bf16(a0, b1, a01);
      a11 = wmma_bf16(a1, b1, a11);
    }
#pragma unroll
    for (int h = 0; h < 2; ++h) {
      const int col = (h ? n1 : n0) + (lane & 15);
      const v8f& q0 = h ? a01 : a00;
      const v8f& q1 = h ? a11 : a10;
      float bb = pp2_b[col], gg = pp2_g[col] * BN_INV, bv = pp2_be[col];
      float mx = -3.0e38f;
#pragma unroll
      for (int v = 0; v < 8; ++v) {
        mx = fmaxf(mx, fmaxf((q0[v] + bb) * gg + bv, 0.0f));
        mx = fmaxf(mx, fmaxf((q1[v] + bb) * gg + bv, 0.0f));
      }
      // lanes L and L+16 hold disjoint row groups of the same column
      mx = fmaxf(mx, __shfl_xor(mx, 16, 32));
      if (lane < 16) xm_pool[(size_t)bi * KM1 + HDIM + col] = (bf16)mx;
    }
  }
}

// ---------------------------------------------------------------------------
// Host side
// ---------------------------------------------------------------------------
extern "C" void kernel_launch(void* const* d_in, const int* in_sizes, int n_in,
                              void* d_out, int out_size, void* d_ws, size_t ws_size,
                              hipStream_t stream) {
  (void)in_sizes; (void)n_in; (void)out_size; (void)ws_size;
  const float* last_pos     = (const float*)d_in[0];
  const float* last_pos_rel = (const float*)d_in[1];
  const float* h0   = (const float*)d_in[2];
  const float* c0   = (const float*)d_in[3];
  /* d_in[4] seq_start_end: uniform scenes, unused */
  const float* emb_W = (const float*)d_in[5];
  const float* emb_b = (const float*)d_in[6];
  const float* lstm_Wih = (const float*)d_in[7];
  const float* lstm_Whh = (const float*)d_in[8];
  const float* lstm_bih = (const float*)d_in[9];
  const float* lstm_bhh = (const float*)d_in[10];
  const float* pos_W = (const float*)d_in[11];
  const float* pos_b = (const float*)d_in[12];
  const float* sp_W  = (const float*)d_in[13];
  const float* sp_b  = (const float*)d_in[14];
  const float* pp1_W = (const float*)d_in[15];
  const float* pp1_b = (const float*)d_in[16];
  const float* pp1_g = (const float*)d_in[17];
  const float* pp1_be = (const float*)d_in[18];
  const float* pp2_W = (const float*)d_in[19];
  const float* pp2_b = (const float*)d_in[20];
  const float* pp2_g = (const float*)d_in[21];
  const float* pp2_be = (const float*)d_in[22];
  const float* m1_W = (const float*)d_in[23];
  const float* m1_b = (const float*)d_in[24];
  const float* m1_g = (const float*)d_in[25];
  const float* m1_be = (const float*)d_in[26];
  const float* m2_W = (const float*)d_in[27];
  const float* m2_b = (const float*)d_in[28];
  const float* m2_g = (const float*)d_in[29];
  const float* m2_be = (const float*)d_in[30];

  char* w = (char*)d_ws;
  auto alloc = [&](size_t bytes) {
    void* p = (void*)w;
    w += (bytes + 255) & ~(size_t)255;
    return p;
  };
  bf16*  wg    = (bf16*)alloc((size_t)GATES * KLSTM * 2);
  float* lbias = (float*)alloc((size_t)GATES * 4);
  bf16*  pp1w  = (bf16*)alloc((size_t)B1 * KLSTM * 2);
  bf16*  pp2w  = (bf16*)alloc((size_t)B2 * B1 * 2);
  bf16*  m1w   = (bf16*)alloc((size_t)M1 * KM1 * 2);
  bf16*  m2w   = (bf16*)alloc((size_t)HDIM * M1 * 2);
  bf16*  xg    = (bf16*)alloc((size_t)NPED * KLSTM * 2);   // [din | h]
  bf16*  xm    = (bf16*)alloc((size_t)NPED * KM1 * 2);     // [h2 | pool]
  float* cbuf  = (float*)alloc((size_t)NPED * HDIM * 4);
  float* curr  = (float*)alloc((size_t)NPED * 2 * 4);
  float* gates = (float*)alloc((size_t)NPED * GATES * 4);
  bf16*  dh1   = (bf16*)alloc((size_t)NPED * M1 * 2);
  float* preds = (float*)d_out;

  // weight prep (runs every call: deterministic, graph-capture safe)
  f2bf_kernel<<<dim3(256), 256, 0, stream>>>(pp1_W, pp1w, B1 * KLSTM);
  f2bf_kernel<<<dim3(1024), 256, 0, stream>>>(pp2_W, pp2w, B2 * B1);
  f2bf_kernel<<<dim3(2048), 256, 0, stream>>>(m1_W, m1w, M1 * KM1);
  f2bf_kernel<<<dim3(512), 256, 0, stream>>>(m2_W, m2w, HDIM * M1);
  build_lstm_w<<<dim3(384), 256, 0, stream>>>(lstm_Wih, lstm_Whh, lstm_bih, lstm_bhh,
                                              wg, lbias);
  init_state<<<dim3(768), 256, 0, stream>>>(last_pos, last_pos_rel, h0, c0, emb_W,
                                            emb_b, xg, cbuf, curr);

  for (int t = 0; t < TSTEPS; ++t) {
    // LSTM gates: [1024x192] x [512x192]^T -> [1024x512] f32
    gemm_bf16<1, false><<<dim3(GATES / 128, NPED / 32), 256, 0, stream>>>(
        xg, KLSTM, wg, KLSTM, lbias, nullptr, nullptr, gates, nullptr, GATES, KLSTM);
    lstm_pointwise<<<dim3(NPED), 128, 0, stream>>>(
        gates, cbuf, xm, xg, curr, preds + (size_t)t * NPED * 2, pos_W, pos_b, emb_W,
        emb_b);
    // fused all-pairs pool net (pp1 -> pp2 -> max over neighbours)
    pool_kernel<<<dim3(NPED), 256, 0, stream>>>(curr, xm, pp1w, pp1_b, pp1_g, pp1_be,
                                                pp2w, pp2_b, pp2_g, pp2_be, sp_W, sp_b,
                                                xm);
    // m1: [1024x1152] x [1024x1152]^T -> relu(bn) -> bf16
    gemm_bf16<2, true><<<dim3(M1 / 128, NPED / 32), 256, 0, stream>>>(
        xm, KM1, m1w, KM1, m1_b, m1_g, m1_be, nullptr, dh1, M1, KM1);
    // m2: [1024x1024] x [128x1024]^T -> relu(bn) -> next h (cols 64..191 of xg)
    gemm_bf16<2, true><<<dim3(HDIM / 128, NPED / 32), 256, 0, stream>>>(
        dh1, M1, m2w, M1, m2_b, m2_g, m2_be, nullptr, xg + EDIM, KLSTM, M1);
  }
}
